// Head_68547678044778
// MI455X (gfx1250) — matrix operand order repaired and checked by reading, not compiled, not run on hardware
//
#include <hip/hip_runtime.h>

typedef __bf16 bf16;
typedef __attribute__((ext_vector_type(16))) __bf16 v16bf;
typedef __attribute__((ext_vector_type(2)))  __bf16 v2bf;
typedef __attribute__((ext_vector_type(8)))  float  v8f;

constexpr int kB = 4, kT = 4096, kE = 1024, kH = 64;
constexpr float kScale = 0.125f;                 // H^-0.5 = 1/8, exact in fp
constexpr float kLog2e = 1.4426950408889634f;

// K-index inside a 16-bit 16x32 A-fragment (ISA 7.12.2), vgpr i, lane-half h
__device__ __forceinline__ int a_k(int i, int h) {
  return ((i < 4) ? 0 : 16) + h * 8 + (i & 3) * 2;
}

// ---------------------------------------------------------------------------
// Kernel 1: weights fp32 [E,H] -> bf16 transposed [3][H][E]
// ---------------------------------------------------------------------------
__global__ void wcvt_kernel(const float* __restrict__ Wq,
                            const float* __restrict__ Wk,
                            const float* __restrict__ Wv,
                            bf16* __restrict__ Wt) {
  int idx = blockIdx.x * blockDim.x + threadIdx.x;
  if (idx >= 3 * kH * kE) return;
  int w = idx / (kH * kE);
  int h = (idx / kE) % kH;
  int e = idx % kE;
  const float* W = (w == 0) ? Wq : ((w == 1) ? Wk : Wv);
  Wt[idx] = (bf16)W[e * kH + h];
}

// ---------------------------------------------------------------------------
// Kernel 2: fused QKV projection. 8 waves/block, 16 rows per wave.
// q,k stored [B*T, H] bf16 (row-major); v stored transposed [B, H, T] bf16.
// ---------------------------------------------------------------------------
__global__ __launch_bounds__(256) void qkv_kernel(
    const float* __restrict__ x, const bf16* __restrict__ Wt,
    bf16* __restrict__ qb, bf16* __restrict__ kbuf, bf16* __restrict__ vt) {
  const int lane = threadIdx.x & 31;
  const int wave = threadIdx.x >> 5;
  const int half = lane >> 4;
  const int n    = lane & 15;
  const int rb   = (blockIdx.x * 8 + wave) * 16;

  v8f acc[12] = {};

  const float* xr = x + (size_t)(rb + n) * kE;
  for (int e = 0; e < kE; e += 32) {
    v16bf a;
#pragma unroll
    for (int i = 0; i < 8; ++i) {
      const int k = a_k(i, half);
      a[2 * i]     = (bf16)xr[e + k];
      a[2 * i + 1] = (bf16)xr[e + k + 1];
    }
#pragma unroll
    for (int f = 0; f < 12; ++f) {
      const bf16* wr = Wt + (size_t)(f * 16 + n) * kE + e;
      v16bf bm;
#pragma unroll
      for (int i = 0; i < 8; ++i) {
        v2bf p = *(const v2bf*)(wr + a_k(i, half));
        bm[2 * i] = p[0]; bm[2 * i + 1] = p[1];
      }
      acc[f] = __builtin_amdgcn_wmma_f32_16x16x32_bf16(
          false, a, false, bm, (short)0, acc[f], false, false);
    }
  }

#pragma unroll
  for (int f = 0; f < 12; ++f) {
    const int w = f >> 2, ht = f & 3;
#pragma unroll
    for (int r = 0; r < 8; ++r) {
      const int row = rb + r + 8 * half;
      const int h   = ht * 16 + n;
      const bf16 val = (bf16)acc[f][r];
      if (w == 0)      qb[(size_t)row * kH + h] = val;
      else if (w == 1) kbuf[(size_t)row * kH + h] = val;
      else {
        const int b = row >> 12;          // / kT
        const int t = row & (kT - 1);
        vt[((size_t)b * kH + h) * kT + t] = val;
      }
    }
  }
}

// ---------------------------------------------------------------------------
// Kernel 3: flash attention. One independent wave per 16 query rows.
// ---------------------------------------------------------------------------
__global__ __launch_bounds__(128) void attn_kernel(
    const bf16* __restrict__ qb, const bf16* __restrict__ kbuf,
    const bf16* __restrict__ vt, float* __restrict__ out) {
  __shared__ bf16 ldsP[4][16 * 32];

  const int lane = threadIdx.x & 31;
  const int wave = threadIdx.x >> 5;
  const int half = lane >> 4;
  const int n    = lane & 15;
  const int g    = blockIdx.x * 4 + wave;   // global 16-row query tile
  const int b    = g >> 8;                  // g / (kT/16)
  const int qrb  = (g & 255) * 16;          // query row base within batch

  // Q fragments: two K=32 chunks over H=64, kept resident
  v16bf aq0, aq1;
  {
    const bf16* qr = qb + (size_t)(b * kT + qrb + n) * kH;
#pragma unroll
    for (int i = 0; i < 8; ++i) {
      const int k = a_k(i, half);
      v2bf p0 = *(const v2bf*)(qr + k);
      v2bf p1 = *(const v2bf*)(qr + 32 + k);
      aq0[2 * i] = p0[0]; aq0[2 * i + 1] = p0[1];
      aq1[2 * i] = p1[0]; aq1[2 * i + 1] = p1[1];
    }
  }

  v8f o[4] = {};
  float mrow[8], lrow[8];
#pragma unroll
  for (int r = 0; r < 8; ++r) { mrow[r] = -__builtin_inff(); lrow[r] = 0.f; }

  const int nchunks = (qrb >> 5) + 1;       // 32 keys per chunk, causal bound
  for (int c = 0; c < nchunks; ++c) {
    const int kbase = c * 32;

    // S tile: 16 queries x 32 keys = two 16x16 WMMA n-tiles, K over H=64
    v8f s0 = {}, s1 = {};
#pragma unroll
    for (int hc = 0; hc < 2; ++hc) {
      const bf16* kr0 = kbuf + (size_t)(b * kT + kbase + n) * kH + hc * 32;
      const bf16* kr1 = kr0 + (size_t)16 * kH;
      v16bf b0, b1;
#pragma unroll
      for (int i = 0; i < 8; ++i) {
        v2bf p0 = *(const v2bf*)(kr0 + a_k(i, half));
        v2bf p1 = *(const v2bf*)(kr1 + a_k(i, half));
        b0[2 * i] = p0[0]; b0[2 * i + 1] = p0[1];
        b1[2 * i] = p1[0]; b1[2 * i + 1] = p1[1];
      }
      const v16bf aq = hc ? aq1 : aq0;
      s0 = __builtin_amdgcn_wmma_f32_16x16x32_bf16(false, aq, false, b0, (short)0, s0, false, false);
      s1 = __builtin_amdgcn_wmma_f32_16x16x32_bf16(false, aq, false, b1, (short)0, s1, false, false);
    }

    const bool lastChunk = (c == nchunks - 1);
#pragma unroll
    for (int r = 0; r < 8; ++r) {
      float v0 = s0[r] * kScale;
      float v1 = s1[r] * kScale;
      if (lastChunk) {                       // causal mask, select keeps EXEC full
        const int qg = qrb + r + 8 * half;
        v0 = (kbase + n      > qg) ? -__builtin_inff() : v0;
        v1 = (kbase + 16 + n > qg) ? -__builtin_inff() : v1;
      }
      float rm = fmaxf(v0, v1);
#pragma unroll
      for (int off = 1; off < 16; off <<= 1)
        rm = fmaxf(rm, __shfl_xor(rm, off, 32));
      const float mnew = fmaxf(mrow[r], rm);
      const float corr = exp2f((mrow[r] - mnew) * kLog2e);
      const float p0 = exp2f((v0 - mnew) * kLog2e);
      const float p1 = exp2f((v1 - mnew) * kLog2e);
      float ps = p0 + p1;
#pragma unroll
      for (int off = 1; off < 16; off <<= 1)
        ps += __shfl_xor(ps, off, 32);
      lrow[r] = lrow[r] * corr + ps;
      mrow[r] = mnew;
#pragma unroll
      for (int ht = 0; ht < 4; ++ht) o[ht][r] *= corr;
      // stage P (C/D layout -> row-major 16x32 tile) for A-fragment reload
      ldsP[wave][(r + 8 * half) * 32 + n]      = (bf16)p0;
      ldsP[wave][(r + 8 * half) * 32 + n + 16] = (bf16)p1;
    }
    asm volatile("s_wait_dscnt 0" ::: "memory");   // wave-local LDS RAW fence

    v16bf ap;
#pragma unroll
    for (int i = 0; i < 8; ++i) {
      const int k = a_k(i, half);
      v2bf p = *(const v2bf*)&ldsP[wave][n * 32 + k];
      ap[2 * i] = p[0]; ap[2 * i + 1] = p[1];
    }

#pragma unroll
    for (int ht = 0; ht < 4; ++ht) {
      const bf16* vr = vt + ((size_t)b * kH + ht * 16 + n) * kT + kbase;
      v16bf bv;
#pragma unroll
      for (int i = 0; i < 8; ++i) {
        v2bf p = *(const v2bf*)(vr + a_k(i, half));
        bv[2 * i] = p[0]; bv[2 * i + 1] = p[1];
      }
      o[ht] = __builtin_amdgcn_wmma_f32_16x16x32_bf16(false, ap, false, bv, (short)0, o[ht], false, false);
    }
  }

#pragma unroll
  for (int r = 0; r < 8; ++r) {
    const float inv = 1.f / lrow[r];
    const int row = b * kT + qrb + r + 8 * half;
#pragma unroll
    for (int ht = 0; ht < 4; ++ht)
      out[(size_t)row * kH + ht * 16 + n] = o[ht][r] * inv;
  }
}

// ---------------------------------------------------------------------------
extern "C" void kernel_launch(void* const* d_in, const int* in_sizes, int n_in,
                              void* d_out, int out_size, void* d_ws, size_t ws_size,
                              hipStream_t stream) {
  // setup_inputs order: x, Wk, Wq, Wv
  const float* x  = (const float*)d_in[0];
  const float* Wk = (const float*)d_in[1];
  const float* Wq = (const float*)d_in[2];
  const float* Wv = (const float*)d_in[3];
  float* out = (float*)d_out;

  char* ws = (char*)d_ws;
  bf16* Wt   = (bf16*)ws;                              // 3*H*E*2  = 384 KB
  bf16* qb   = (bf16*)(ws + 3 * kH * kE * 2);          // B*T*H*2  = 2 MB
  bf16* kbuf = qb + (size_t)kB * kT * kH;              // 2 MB
  bf16* vt   = kbuf + (size_t)kB * kT * kH;            // 2 MB (transposed [B,H,T])

  wcvt_kernel<<<(3 * kH * kE + 255) / 256, 256, 0, stream>>>(Wq, Wk, Wv, Wt);
  qkv_kernel<<<kB * kT / 128, 256, 0, stream>>>(x, Wt, qb, kbuf, vt);
  attn_kernel<<<kB * kT / 64, 128, 0, stream>>>(qb, kbuf, vt, out);
}
